// SimVQ_10428180595128
// MI455X (gfx1250) — matrix operand order repaired and checked by reading, not compiled, not run on hardware
//
#include <hip/hip_runtime.h>
#include <math.h>

// ---------------- problem constants (from setup_inputs) ----------------
#define N_E    16384      // codebook entries
#define EDIM   256        // e_dim == c
#define NTOK   8192       // b*h*w = 8*32*32
#define HW     1024       // h*w
#define EPS_   1e-6f
#define NTILE  (N_E / 16) // 1024 code tiles

// z_q floats | loss | indices(as float)
#define OUT_LOSS_OFF   2097152
#define OUT_IDX_OFF    2097153

typedef __attribute__((ext_vector_type(16))) __bf16        v16bf;
typedef __attribute__((ext_vector_type(8)))  float         v8f;
typedef __attribute__((ext_vector_type(4)))  unsigned int  u32x4;
typedef __attribute__((ext_vector_type(8)))  int           i32x8;
typedef __attribute__((ext_vector_type(4)))  int           i32x4;

union V16BF { v16bf v; __bf16 e[16]; };
union V8F   { v8f   v; float  e[8];  };

// CDNA5 ISA: 16-bit A-matrix 16x32 fragment K mapping.
// lane half h (0: lanes0-15, 1: lanes16-31), element e (0..15):
//   pairs 0..3 -> K = 8h + 2p + w ; pairs 4..7 -> K = 16 + 8h + 2(p-4) + w
__device__ __forceinline__ int kmapA(int e, int half) {
  int p = e >> 1, w = e & 1;
  return (p < 4) ? (half * 8 + p * 2 + w) : (16 + half * 8 + (p - 4) * 2 + w);
}

__device__ __forceinline__ float waveSum(float v) {
  #pragma unroll
  for (int o = 16; o > 0; o >>= 1) v += __shfl_xor(v, o, 32);
  return v;
}

// ---------------------------------------------------------------------------
// TDM: DMA one 16-row x 256-col bf16 tile (8 KB) from global emb into LDS.
// D# per CDNA5 ISA:
//   group0: count=1 | lds_addr | global_addr[56:0] | type=2
//   group1: data_size=1 (2B), tensor_dim0=256, tensor_dim1=16,
//           tile_dim0=256, tile_dim1=16, tensor_dim0_stride=256
// Tracked by TENSORcnt; issued by one wave, completion published via barrier.
// ---------------------------------------------------------------------------
__device__ __forceinline__ void tdm_load_tile(const __bf16* gsrc,
                                              unsigned lds_off) {
  unsigned long long ga = (unsigned long long)(uintptr_t)gsrc;
  u32x4 g0;
  g0[0] = 1u;                                            // count=1, user D#
  g0[1] = lds_off;                                       // lds_addr (bytes)
  g0[2] = (unsigned)(ga & 0xFFFFFFFFu);                  // global_addr lo
  g0[3] = (unsigned)((ga >> 32) & 0x01FFFFFFu) | (2u << 30);  // hi | type=2
  i32x8 g1;
  g1[0] = (int)(1u << 16);        // workgroup_mask=0, data_size=1 (2 bytes)
  g1[1] = (int)(256u << 16);      // tensor_dim0 = 256 (bits 63:48)
  g1[2] = (int)(16u << 16);       // tensor_dim1 = 16  (bits 95:80)
  g1[3] = (int)(256u << 16);      // tile_dim0   = 256 (bits 127:112)
  g1[4] = 16;                     // tile_dim1   = 16  (bits 143:128)
  g1[5] = 256;                    // tensor_dim0_stride = 256 (bits 207:160)
  g1[6] = 0;
  g1[7] = 0;
  i32x4 z4 = {0, 0, 0, 0};
#if __clang_major__ >= 23
  i32x8 z8 = {0, 0, 0, 0, 0, 0, 0, 0};
  __builtin_amdgcn_tensor_load_to_lds(g0, g1, z4, z4, z8, 0);
#else
  __builtin_amdgcn_tensor_load_to_lds(g0, g1, z4, z4, 0);
#endif
}

// ---------------------------------------------------------------------------
// Kernel 1: emb = codebook @ W^T  (16384x256, K=256), one wave per 16x16 tile.
// D accum is f32; store both f32 (gather/rotation path) and bf16 (dist GEMM).
// ---------------------------------------------------------------------------
__global__ __launch_bounds__(32)
void emb_gemm_kernel(const float* __restrict__ codebook,
                     const float* __restrict__ Wm,
                     float* __restrict__ emb_f32,
                     __bf16* __restrict__ emb_bf16) {
  const int lane = threadIdx.x;
  const int nloc = lane & 15;
  const int half = lane >> 4;
  const int i0 = blockIdx.x * 16;   // codebook row tile (M)
  const int j0 = blockIdx.y * 16;   // output channel tile (N)

  v8f acc = {};
  #pragma unroll
  for (int k0 = 0; k0 < EDIM; k0 += 32) {
    V16BF a, b;
    const float* arow = codebook + (size_t)(i0 + nloc) * EDIM + k0;
    // B[k][n] = W[n][k]; lane holds column n, 16 consecutive K per half.
    const float* brow = Wm + (size_t)(j0 + nloc) * EDIM + k0 + half * 16;
    #pragma unroll
    for (int e = 0; e < 16; ++e) {
      a.e[e] = (__bf16)arow[half * 16 + e];
      b.e[e] = (__bf16)brow[e];
    }
    acc = __builtin_amdgcn_wmma_f32_16x16x32_bf16(false, a.v, false, b.v,
                                                  (short)0, acc, false, false);
  }
  V8F d; d.v = acc;
  #pragma unroll
  for (int r = 0; r < 8; ++r) {
    const int m   = r + 8 * half;          // D row = vgpr + 8*(lane/16)
    const int row = i0 + m, col = j0 + nloc;
    emb_f32 [(size_t)row * EDIM + col] = d.e[r];
    emb_bf16[(size_t)row * EDIM + col] = (__bf16)d.e[r];
  }
}

// ---------------------------------------------------------------------------
// Kernel 2: emb_norm[i] = sum_j emb[i][j]^2  (one block per row)
// ---------------------------------------------------------------------------
__global__ __launch_bounds__(256)
void emb_norm_kernel(const float* __restrict__ emb_f32,
                     float* __restrict__ emb_norm) {
  const int row = blockIdx.x;
  const float v = emb_f32[(size_t)row * EDIM + threadIdx.x];
  float s = waveSum(v * v);
  __shared__ float part[8];
  const int lane = threadIdx.x & 31, wv = threadIdx.x >> 5;
  if (lane == 0) part[wv] = s;
  __syncthreads();
  if (threadIdx.x == 0) {
    float tot = 0.f;
    #pragma unroll
    for (int i = 0; i < 8; ++i) tot += part[i];
    emb_norm[row] = tot;
  }
}

// ---------------------------------------------------------------------------
// Kernel 3: streaming distance GEMM + argmin.
// Block = 64 tokens (4 waves x one 16-token tile). Every code tile (16x256
// bf16 = 8 KB) is staged ONCE per block into LDS by the Tensor Data Mover,
// double-buffered and overlapped with WMMA compute; all 4 waves consume it
// (4x less L2 codebook traffic than per-wave streaming: 128 blocks x 8 MB).
// argmin key = ||emb||^2 - 2*dot (||z||^2 constant per row); first-index
// tie-break like jnp.argmin.
// ---------------------------------------------------------------------------
__global__ __launch_bounds__(128)
void argmin_kernel(const float* __restrict__ z,
                   const __bf16* __restrict__ emb_bf16,
                   const float* __restrict__ emb_norm,
                   int*   __restrict__ indices,
                   float* __restrict__ out_idx_f) {
  const int lane = threadIdx.x & 31;
  const int wv   = threadIdx.x >> 5;
  const int nloc = lane & 15;
  const int half = lane >> 4;
  const int tw   = blockIdx.x * 4 + wv;   // this wave's 16-token tile

  __shared__ __align__(64) __bf16 lbuf[2][16 * EDIM];   // 2 x 8 KB

  // ---- A fragments: z tile 16 tokens x 256 channels, strided gather from
  //      (b,c,h,w): z_flat[t][c] = z[b*256*1024 + c*1024 + hw]
  const int tok  = tw * 16 + nloc;
  const int bidx = tok >> 10;
  const int hw   = tok & (HW - 1);
  const float* zbase = z + (size_t)bidx * EDIM * HW + hw;
  V16BF afr[8];
  #pragma unroll
  for (int kk = 0; kk < 8; ++kk) {
    #pragma unroll
    for (int e = 0; e < 16; ++e) {
      const int ch = kk * 32 + half * 16 + e;
      afr[kk].e[e] = (__bf16)zbase[(size_t)ch * HW];
    }
  }

  float bestVal[8];
  int   bestIdx[8];
  #pragma unroll
  for (int r = 0; r < 8; ++r) { bestVal[r] = 3.4e38f; bestIdx[r] = 0x7fffffff; }

  const unsigned lds0 = (unsigned)(uintptr_t)(&lbuf[0][0]);
  const unsigned lds1 = (unsigned)(uintptr_t)(&lbuf[1][0]);

  // preload tile 0
  if (threadIdx.x < 32) {
    tdm_load_tile(emb_bf16, lds0);
    __builtin_amdgcn_s_wait_tensorcnt(0);
  }
  __syncthreads();

  for (int nt = 0; nt < NTILE; ++nt) {
    const int cur = nt & 1;
    // kick off DMA of the next tile into the other buffer (wave 0 only;
    // TDM ignores EXEC and is tracked per-wave via TENSORcnt)
    if (threadIdx.x < 32) {
      if (nt + 1 < NTILE)
        tdm_load_tile(emb_bf16 + (size_t)(nt + 1) * 16 * EDIM,
                      cur ? lds0 : lds1);
    }

    // B[k][n] = emb[n][k]: lane = column n, 16 contiguous bf16 per half
    const __bf16* brow = &lbuf[cur][nloc * EDIM + half * 16];
    v8f acc = {};
    #pragma unroll
    for (int kk = 0; kk < 8; ++kk) {
      v16bf bfr = *(const v16bf*)(brow + kk * 32);
      acc = __builtin_amdgcn_wmma_f32_16x16x32_bf16(false, afr[kk].v, false,
                                                    bfr, (short)0, acc,
                                                    false, false);
    }
    const int   n  = nt * 16 + nloc;
    const float en = emb_norm[n];
    V8F d; d.v = acc;
    #pragma unroll
    for (int r = 0; r < 8; ++r) {
      const float dist = en - 2.0f * d.e[r];
      if (dist < bestVal[r]) { bestVal[r] = dist; bestIdx[r] = n; }
    }

    // next tile landed? then publish it to all 4 waves
    if (threadIdx.x < 32) __builtin_amdgcn_s_wait_tensorcnt(0);
    __syncthreads();
  }

  // cross-lane argmin within each 16-lane half (xor masks stay in-half)
  #pragma unroll
  for (int off = 1; off < 16; off <<= 1) {
    #pragma unroll
    for (int r = 0; r < 8; ++r) {
      const float ov = __shfl_xor(bestVal[r], off, 32);
      const int   oi = __shfl_xor(bestIdx[r], off, 32);
      if (ov < bestVal[r] || (ov == bestVal[r] && oi < bestIdx[r])) {
        bestVal[r] = ov; bestIdx[r] = oi;
      }
    }
  }
  // lane 0 holds rows 0..7, lane 16 holds rows 8..15
  if (nloc == 0) {
    #pragma unroll
    for (int r = 0; r < 8; ++r) {
      const int t2 = tw * 16 + (r + 8 * half);
      indices[t2]   = bestIdx[r];
      out_idx_f[t2] = (float)bestIdx[r];
    }
  }
}

// ---------------------------------------------------------------------------
// Kernel 4: zero the loss accumulator (d_out is poisoned before timing)
// ---------------------------------------------------------------------------
__global__ void init_loss_kernel(float* p) { *p = 0.0f; }

// ---------------------------------------------------------------------------
// Kernel 5: gather + loss + rotation trick (forward), one block per token.
//   loss = COMMIT_W*(mse + BETA*mse) = 1.25 * mean((z - z_q)^2)
//   rot  = src - 2(src.w)w + 2(src.u)q, scaled by |tgt|/(|src|+eps);
//   all scalars derive from ss=|src|^2, tt=|tgt|^2, st=src.tgt.
// ---------------------------------------------------------------------------
__global__ __launch_bounds__(256)
void rotate_out_kernel(const float* __restrict__ z,
                       const float* __restrict__ emb_f32,
                       const int*   __restrict__ indices,
                       float* __restrict__ out,
                       float* __restrict__ loss_ptr) {
  const int t    = blockIdx.x;
  const int c    = threadIdx.x;
  const int bidx = t >> 10;
  const int hw   = t & (HW - 1);
  const int idx  = indices[t];

  const float src = z[(size_t)bidx * EDIM * HW + (size_t)c * HW + hw];
  const float tgt = emb_f32[(size_t)idx * EDIM + c];

  float ss = waveSum(src * src);
  float tt = waveSum(tgt * tgt);
  float st = waveSum(src * tgt);

  __shared__ float lss[8], ltt[8], lst[8];
  const int lane = c & 31, wv = c >> 5;
  if (lane == 0) { lss[wv] = ss; ltt[wv] = tt; lst[wv] = st; }
  __syncthreads();
  if (c == 0) {
    float a = 0.f, b = 0.f, d = 0.f;
    #pragma unroll
    for (int i = 0; i < 8; ++i) { a += lss[i]; b += ltt[i]; d += lst[i]; }
    lss[0] = a; ltt[0] = b; lst[0] = d;
    const float sd = a - 2.0f * d + b;   // sum (src - tgt)^2
    atomicAdd(loss_ptr, sd * (1.25f / ((float)NTOK * (float)EDIM)));
  }
  __syncthreads();
  ss = lss[0]; tt = ltt[0]; st = lst[0];

  const float ns    = sqrtf(ss), ntg = sqrtf(tt);
  const float inv_s = 1.0f / (ns + EPS_);
  const float inv_t = 1.0f / (ntg + EPS_);
  const float uu = ss * inv_s * inv_s;
  const float qq = tt * inv_t * inv_t;
  const float uq = st * inv_s * inv_t;
  const float wn   = sqrtf(uu + 2.0f * uq + qq);
  const float invw = 1.0f / (wn + EPS_);
  const float dw = (ss * inv_s + st * inv_t) * invw;  // src . w_
  const float du = ss * inv_s;                        // src . u
  const float wc = (src * inv_s + tgt * inv_t) * invw;
  const float qc = tgt * inv_t;
  const float rot = src - 2.0f * dw * wc + 2.0f * du * qc;

  out[(size_t)bidx * EDIM * HW + (size_t)c * HW + hw] = rot * (ntg * inv_s);
}

// ---------------------------------------------------------------------------
extern "C" void kernel_launch(void* const* d_in, const int* in_sizes, int n_in,
                              void* d_out, int out_size, void* d_ws, size_t ws_size,
                              hipStream_t stream) {
  (void)in_sizes; (void)n_in; (void)out_size; (void)ws_size;
  const float* z        = (const float*)d_in[0];  // (8,256,32,32)
  const float* codebook = (const float*)d_in[1];  // (16384,256)
  const float* Wm       = (const float*)d_in[2];  // (256,256)
  float* out = (float*)d_out;

  // workspace layout
  char* ws = (char*)d_ws;
  __bf16* emb_bf16 = (__bf16*)ws;                                   //  8 MB
  size_t off = (size_t)N_E * EDIM * sizeof(__bf16);
  float* emb_f32  = (float*)(ws + off);                             // 16 MB
  off += (size_t)N_E * EDIM * sizeof(float);
  float* emb_norm = (float*)(ws + off);                             // 64 KB
  off += (size_t)N_E * sizeof(float);
  int*   indices  = (int*)(ws + off);                               // 32 KB

  float* loss_ptr = out + OUT_LOSS_OFF;
  float* out_idx  = out + OUT_IDX_OFF;

  emb_gemm_kernel<<<dim3(N_E / 16, EDIM / 16), 32, 0, stream>>>(
      codebook, Wm, emb_f32, emb_bf16);
  emb_norm_kernel<<<N_E, 256, 0, stream>>>(emb_f32, emb_norm);
  argmin_kernel<<<NTOK / 64, 128, 0, stream>>>(
      z, emb_bf16, emb_norm, indices, out_idx);
  init_loss_kernel<<<1, 1, 0, stream>>>(loss_ptr);
  rotate_out_kernel<<<NTOK, 256, 0, stream>>>(
      z, emb_f32, indices, out, loss_ptr);
}
